// Model_88201448391442
// MI455X (gfx1250) — compile-verified, matrix operand order checked by reading, not run on hardware
//
#include <hip/hip_runtime.h>

#define N_NODES 50000
#define N_EDGES 800000
#define F_IN 128
#define HID 128
#define EMB 128
#define SEQ 20
#define LSTM_H 256
#define BATCH 1024
#define GATES (4 * LSTM_H)
#define BN_EPS 1e-5f
#define HEAD_NT 5   /* 5 x 16 = 80 cols per wave; 50000 = 625 * 80 */

typedef __attribute__((ext_vector_type(16))) __bf16 v16bf;
typedef __attribute__((ext_vector_type(8)))  __bf16 v8bf;
typedef __attribute__((ext_vector_type(8)))  float  v8f;
typedef __attribute__((ext_vector_type(2)))  __bf16 bf16x2;

// ---------------------------------------------------------------------------
// WMMA fragment loader (global).
// 16-bit A (16x32 MxK) layout (ISA 7.12.2): lane L -> row = L%16, half = L/16;
// VGPR v packs K = (v&3)*2 + {0,1} + half*8 + (v>>2)*16.  B for C = A * W^T
// uses the identical mapping with "row" = output column (W row-major [N,K]).
// Per lane this is two contiguous 16B runs at +0 and +32 bytes -> 2x b128.
// ---------------------------------------------------------------------------
static __device__ __forceinline__ v16bf load_frag(const __bf16* base, int ld, int k0) {
  const int lane = threadIdx.x & 31;
  const int r    = lane & 15;
  const int half = lane >> 4;
  const __bf16* p = base + (size_t)r * ld + k0 + half * 8;
  v16bf out;
#pragma unroll
  for (int v = 0; v < 8; ++v) {
    const int kb = (v & 3) * 2 + (v >> 2) * 16;
    bf16x2 pr = *reinterpret_cast<const bf16x2*>(p + kb);
    out[2 * v]     = pr[0];
    out[2 * v + 1] = pr[1];
  }
  return out;
}

static __device__ __forceinline__ float sigmf(float x) {
  return 1.0f / (1.0f + __expf(-x));
}

// ---------------------------------------------------------------------------
// Generic direct-load WMMA GEMM: C[M,N] = A[M,K] * B[N,K]^T, bf16 in / f32 out.
// One wave computes a 16 x (16*NT) tile: A fragment reused NT times.
// ---------------------------------------------------------------------------
template <int NT, int EPI>
__global__ __launch_bounds__(32) void k_gemm_bt(const __bf16* __restrict__ A, int lda,
                                                const __bf16* __restrict__ B, int ldb,
                                                const float* __restrict__ bias,
                                                float* __restrict__ C, int ldc, int K) {
  const int row0 = blockIdx.x * 16;
  const int col0 = blockIdx.y * (16 * NT);
  const __bf16* Abase = A + (size_t)row0 * lda;
  const int lane = threadIdx.x & 31;
  v8f acc[NT] = {};
  for (int k0 = 0; k0 < K; k0 += 32) {
    if (k0 + 32 < K)
      __builtin_prefetch(Abase + (size_t)(lane & 15) * lda + k0 + 32, 0, 3);
    v16bf a = load_frag(Abase, lda, k0);
#pragma unroll
    for (int q = 0; q < NT; ++q) {
      v16bf b = load_frag(B + (size_t)(col0 + q * 16) * ldb, ldb, k0);
      acc[q] = __builtin_amdgcn_wmma_f32_16x16x32_bf16(false, a, false, b, (short)0,
                                                       acc[q], false, false);
    }
  }
  // C layout: lane -> n = lane&15 ; VGPR r -> m = r + 8*(lane>>4)
  const int n  = lane & 15;
  const int mb = (lane >> 4) * 8;
#pragma unroll
  for (int q = 0; q < NT; ++q) {
    const float bv = (EPI == 1) ? bias[col0 + q * 16 + n] : 0.0f;
#pragma unroll
    for (int r = 0; r < 8; ++r)
      C[(size_t)(row0 + mb + r) * ldc + col0 + q * 16 + n] = acc[q][r] + bv;
  }
}

// ---------------------------------------------------------------------------
// Prediction head GEMM with CDNA5 async-LDS staging of B (pred_w) tiles.
// Block = 4 waves (128 thr); wave w owns batch rows m0+w*16; all waves share
// HEAD_NT staged 16x32 B tiles per k-chunk (B traffic / 4).  Staging uses
// global_load_async_to_lds_b128 (ASYNCcnt) + s_wait_asynccnt, consumption via
// ds loads feeding v_wmma.  Tile row = 64B: frag = two 16B runs (+0 / +32).
// ---------------------------------------------------------------------------
__global__ __launch_bounds__(128) void k_gemm_head(const __bf16* __restrict__ A,
                                                   const __bf16* __restrict__ B,
                                                   const float* __restrict__ bias,
                                                   float* __restrict__ C) {
  __shared__ __align__(16) char sB[HEAD_NT * 1024];
  const int tid  = threadIdx.x;
  const int wave = tid >> 5;
  const int lane = tid & 31;
  const int m0   = blockIdx.x * 64 + wave * 16;
  const int col0 = blockIdx.y * (HEAD_NT * 16);
  const __bf16* Abase = A + (size_t)m0 * LSTM_H;
  v8f acc[HEAD_NT] = {};

  for (int ks = 0; ks < LSTM_H / 32; ++ks) {
    const int k0 = ks * 32;
    // ---- async-stage HEAD_NT B tiles (16 rows x 64B each) into LDS ----
    for (int u = tid; u < HEAD_NT * 64; u += 128) {   // 16B units
      const int q  = u >> 6;
      const int r  = (u >> 2) & 15;
      const int cc = u & 3;
      const __bf16* g = B + (size_t)(col0 + q * 16 + r) * LSTM_H + k0 + cc * 8;
      unsigned lo = (unsigned)(uintptr_t)&sB[q * 1024 + r * 64 + cc * 16];
      asm volatile("global_load_async_to_lds_b128 %0, %1, off"
                   :: "v"(lo), "v"(g) : "memory");
    }
    asm volatile("s_wait_asynccnt 0x0" ::: "memory");
    __syncthreads();   // staged data from all waves now visible

    v16bf a = load_frag(Abase, LSTM_H, k0);
    const int fr = lane & 15, half = lane >> 4;
#pragma unroll
    for (int q = 0; q < HEAD_NT; ++q) {
      const char* p = &sB[q * 1024 + fr * 64 + half * 16];
      v8bf blo = *reinterpret_cast<const v8bf*>(p);
      v8bf bhi = *reinterpret_cast<const v8bf*>(p + 32);
      v16bf b;
#pragma unroll
      for (int i = 0; i < 8; ++i) { b[i] = blo[i]; b[8 + i] = bhi[i]; }
      acc[q] = __builtin_amdgcn_wmma_f32_16x16x32_bf16(false, a, false, b, (short)0,
                                                       acc[q], false, false);
    }
    __syncthreads();   // all waves done reading before restaging
  }

  const int n  = lane & 15;
  const int mb = (lane >> 4) * 8;
#pragma unroll
  for (int q = 0; q < HEAD_NT; ++q) {
    const float bv = bias[col0 + q * 16 + n];
#pragma unroll
    for (int r = 0; r < 8; ++r)
      C[(size_t)(m0 + mb + r) * N_NODES + col0 + q * 16 + n] = acc[q][r] + bv;
  }
}

// ---------------------------------------------------------------------------
// LSTM step: gates = x_t @ Wih^T + h @ Whh^T + b; one wave per 16-batch x
// 16-hidden tile holding 4 gate accumulators (A fragment reused 4x).
// ---------------------------------------------------------------------------
__global__ __launch_bounds__(32) void k_lstm_step(int t,
                                                  const __bf16* __restrict__ paths_bf,
                                                  const __bf16* __restrict__ w_ih,
                                                  const __bf16* __restrict__ w_hh,
                                                  const float* __restrict__ b_ih,
                                                  const float* __restrict__ b_hh,
                                                  const __bf16* __restrict__ h_in,
                                                  float* __restrict__ c,
                                                  float* __restrict__ h_f32,
                                                  __bf16* __restrict__ h_out) {
  const int m0 = blockIdx.x * 16;
  const int j0 = blockIdx.y * 16;
  v8f acc[4] = {};

  const __bf16* Ax = paths_bf + (size_t)m0 * (SEQ * EMB) + (size_t)t * EMB;
#pragma unroll
  for (int k0 = 0; k0 < EMB; k0 += 32) {
    v16bf a = load_frag(Ax, SEQ * EMB, k0);
#pragma unroll
    for (int q = 0; q < 4; ++q) {
      v16bf b = load_frag(w_ih + (size_t)(q * LSTM_H + j0) * EMB, EMB, k0);
      acc[q] = __builtin_amdgcn_wmma_f32_16x16x32_bf16(false, a, false, b, (short)0,
                                                       acc[q], false, false);
    }
  }
  const __bf16* Ah = h_in + (size_t)m0 * LSTM_H;
#pragma unroll
  for (int k0 = 0; k0 < LSTM_H; k0 += 32) {
    v16bf a = load_frag(Ah, LSTM_H, k0);
#pragma unroll
    for (int q = 0; q < 4; ++q) {
      v16bf b = load_frag(w_hh + (size_t)(q * LSTM_H + j0) * LSTM_H, LSTM_H, k0);
      acc[q] = __builtin_amdgcn_wmma_f32_16x16x32_bf16(false, a, false, b, (short)0,
                                                       acc[q], false, false);
    }
  }

  const int lane = threadIdx.x & 31;
  const int n    = lane & 15;
  const int half = lane >> 4;
  const int j    = j0 + n;
  const float bi = b_ih[j] + b_hh[j];
  const float bf = b_ih[j + LSTM_H] + b_hh[j + LSTM_H];
  const float bg = b_ih[j + 2 * LSTM_H] + b_hh[j + 2 * LSTM_H];
  const float bo = b_ih[j + 3 * LSTM_H] + b_hh[j + 3 * LSTM_H];
#pragma unroll
  for (int r = 0; r < 8; ++r) {
    const int m = m0 + r + 8 * half;
    const size_t o = (size_t)m * LSTM_H + j;
    const float gi = acc[0][r] + bi;
    const float gf = acc[1][r] + bf;
    const float gg = acc[2][r] + bg;
    const float go = acc[3][r] + bo;
    const float cn = sigmf(gf) * c[o] + sigmf(gi) * tanhf(gg);
    const float hn = sigmf(go) * tanhf(cn);
    c[o] = cn;
    h_f32[o] = hn;
    h_out[o] = (__bf16)hn;
  }
}

// ------------------------- elementwise / graph kernels ----------------------
__global__ void k_deg(const int* __restrict__ dst, float* __restrict__ deg) {
  int i = blockIdx.x * blockDim.x + threadIdx.x;
  if (i < N_EDGES) atomicAdd(&deg[dst[i]], 1.0f);
}

__global__ void k_dinv(float* __restrict__ deg) {
  int i = blockIdx.x * blockDim.x + threadIdx.x;
  if (i < N_NODES) deg[i] = rsqrtf(deg[i] + 1.0f);  // +1 = self-loop
}

__global__ void k_f32_to_bf16(const float* __restrict__ in, __bf16* __restrict__ out,
                              size_t n) {
  size_t i = (size_t)blockIdx.x * blockDim.x + threadIdx.x;
  if (i < n) out[i] = (__bf16)in[i];
}

__global__ void k_transpose_bf16(const float* __restrict__ W, __bf16* __restrict__ Wt,
                                 int R, int C) {
  int i = blockIdx.x * blockDim.x + threadIdx.x;
  if (i < R * C) {
    int r = i / C, cidx = i % C;
    Wt[(size_t)cidx * R + r] = (__bf16)W[(size_t)r * C + cidx];
  }
}

__global__ void k_scale_rows(float* __restrict__ H, const float* __restrict__ dinv) {
  int i = blockIdx.x * blockDim.x + threadIdx.x;  // N_NODES*128 threads
  H[i] *= dinv[i >> 7];
}

__global__ void k_scatter(const int* __restrict__ src, const int* __restrict__ dst,
                          const float* __restrict__ HS, float* __restrict__ AGG) {
  long long tid = (long long)blockIdx.x * blockDim.x + threadIdx.x;
  int e  = (int)(tid >> 5);
  int fb = ((int)tid & 31) * 4;
  int s = src[e], d = dst[e];
  const float* hp = HS + (size_t)s * 128 + fb;
  float* ap = AGG + (size_t)d * 128 + fb;
#pragma unroll
  for (int j = 0; j < 4; ++j) atomicAdd(ap + j, hp[j]);
}

__global__ void k_gcn_fin_relu_bf(const float* __restrict__ AGG,
                                  const float* __restrict__ HS,
                                  const float* __restrict__ dinv,
                                  const float* __restrict__ bias,
                                  __bf16* __restrict__ out) {
  int i = blockIdx.x * blockDim.x + threadIdx.x;
  float v = dinv[i >> 7] * (AGG[i] + HS[i]) + bias[i & 127];
  out[i] = (__bf16)fmaxf(v, 0.0f);
}

__global__ void k_gcn_fin_f32(const float* __restrict__ AGG, float* __restrict__ HS,
                              const float* __restrict__ dinv,
                              const float* __restrict__ bias) {
  int i = blockIdx.x * blockDim.x + threadIdx.x;
  HS[i] = dinv[i >> 7] * (AGG[i] + HS[i]) + bias[i & 127];
}

__global__ void k_gather(const int* __restrict__ idx, const float* __restrict__ emb,
                         float* __restrict__ paths) {
  int i = blockIdx.x * blockDim.x + threadIdx.x;  // BATCH*SEQ*EMB threads
  int b = i / (SEQ * EMB);
  int rem = i % (SEQ * EMB);
  int t = rem >> 7, f = rem & 127;
  int id = idx[b * SEQ + t];
  paths[i] = (id < N_NODES) ? emb[(size_t)id * EMB + f] : 0.0f;
}

__global__ void k_bn1_stats(const float* __restrict__ paths, float* __restrict__ mu,
                            float* __restrict__ rstd) {
  __shared__ float ss[256], sq[256];
  const int t = blockIdx.x;
  float s = 0.f, q = 0.f;
  for (int i = threadIdx.x; i < BATCH * EMB; i += 256) {
    int b = i >> 7, f = i & 127;
    float v = paths[(size_t)b * (SEQ * EMB) + t * EMB + f];
    s += v;
    q += v * v;
  }
  ss[threadIdx.x] = s; sq[threadIdx.x] = q; __syncthreads();
  for (int st = 128; st > 0; st >>= 1) {
    if (threadIdx.x < st) { ss[threadIdx.x] += ss[threadIdx.x + st];
                            sq[threadIdx.x] += sq[threadIdx.x + st]; }
    __syncthreads();
  }
  if (threadIdx.x == 0) {
    const float inv_n = 1.0f / (BATCH * EMB);
    float m = ss[0] * inv_n;
    float var = sq[0] * inv_n - m * m;
    mu[t] = m;
    rstd[t] = rsqrtf(var + BN_EPS);
  }
}

__global__ void k_bn1_norm(const float* __restrict__ paths, const float* __restrict__ mu,
                           const float* __restrict__ rstd, const float* __restrict__ g,
                           const float* __restrict__ b, __bf16* __restrict__ out) {
  int i = blockIdx.x * blockDim.x + threadIdx.x;
  int t = (i >> 7) % SEQ;
  out[i] = (__bf16)((paths[i] - mu[t]) * rstd[t] * g[t] + b[t]);
}

__global__ void k_bn2_stats(const float* __restrict__ h, float* __restrict__ mu,
                            float* __restrict__ rstd) {
  __shared__ float ss[256], sq[256];
  const int f = blockIdx.x;
  float s = 0.f, q = 0.f;
  for (int b = threadIdx.x; b < BATCH; b += 256) {
    float v = h[(size_t)b * LSTM_H + f];
    s += v; q += v * v;
  }
  ss[threadIdx.x] = s; sq[threadIdx.x] = q; __syncthreads();
  for (int st = 128; st > 0; st >>= 1) {
    if (threadIdx.x < st) { ss[threadIdx.x] += ss[threadIdx.x + st];
                            sq[threadIdx.x] += sq[threadIdx.x + st]; }
    __syncthreads();
  }
  if (threadIdx.x == 0) {
    float m = ss[0] / BATCH;
    float var = sq[0] / BATCH - m * m;
    mu[f] = m;
    rstd[f] = rsqrtf(var + BN_EPS);
  }
}

__global__ void k_bn2_norm(const float* __restrict__ h, const float* __restrict__ mu,
                           const float* __restrict__ rstd, const float* __restrict__ g,
                           const float* __restrict__ b, __bf16* __restrict__ out) {
  int i = blockIdx.x * blockDim.x + threadIdx.x;  // BATCH*LSTM_H threads
  int f = i & (LSTM_H - 1);
  out[i] = (__bf16)((h[i] - mu[f]) * rstd[f] * g[f] + b[f]);
}

__global__ void k_row_logZ(const float* __restrict__ logits, float* __restrict__ logZ) {
  __shared__ float s[256];
  const int r = blockIdx.x;
  const float* row = logits + (size_t)r * N_NODES;
  float mx = -3.4e38f;
  for (int i = threadIdx.x; i < N_NODES; i += 256) mx = fmaxf(mx, row[i]);
  s[threadIdx.x] = mx; __syncthreads();
  for (int st = 128; st > 0; st >>= 1) {
    if (threadIdx.x < st) s[threadIdx.x] = fmaxf(s[threadIdx.x], s[threadIdx.x + st]);
    __syncthreads();
  }
  mx = s[0]; __syncthreads();
  float sum = 0.f;
  for (int i = threadIdx.x; i < N_NODES; i += 256) sum += __expf(row[i] - mx);
  s[threadIdx.x] = sum; __syncthreads();
  for (int st = 128; st > 0; st >>= 1) {
    if (threadIdx.x < st) s[threadIdx.x] += s[threadIdx.x + st];
    __syncthreads();
  }
  if (threadIdx.x == 0) logZ[r] = mx + __logf(s[0]);
}

__global__ void k_sub_logZ(float* __restrict__ out, const float* __restrict__ logZ) {
  int i = blockIdx.x * blockDim.x + threadIdx.x;  // BATCH*N_NODES threads
  out[i] -= logZ[i / N_NODES];
}

// ---------------------------------------------------------------------------
extern "C" void kernel_launch(void* const* d_in, const int* in_sizes, int n_in,
                              void* d_out, int out_size, void* d_ws, size_t ws_size,
                              hipStream_t stream) {
  const int*   indices = (const int*)d_in[0];
  const float* graph_x = (const float*)d_in[1];
  const int*   e_src   = (const int*)d_in[2];
  const int*   e_dst   = e_src + N_EDGES;
  const float* w0    = (const float*)d_in[3];
  const float* b0    = (const float*)d_in[4];
  const float* w1    = (const float*)d_in[5];
  const float* b1    = (const float*)d_in[6];
  const float* bn1_g = (const float*)d_in[7];
  const float* bn1_b = (const float*)d_in[8];
  const float* w_ih  = (const float*)d_in[9];
  const float* w_hh  = (const float*)d_in[10];
  const float* b_ih  = (const float*)d_in[11];
  const float* b_hh  = (const float*)d_in[12];
  const float* bn2_g = (const float*)d_in[13];
  const float* bn2_b = (const float*)d_in[14];
  const float* pred_w = (const float*)d_in[15];
  const float* pred_b = (const float*)d_in[16];
  float* out = (float*)d_out;
  (void)in_sizes; (void)n_in; (void)out_size; (void)ws_size;

  size_t off = 0;
  auto alloc = [&](size_t bytes) -> void* {
    off = (off + 255) & ~(size_t)255;
    void* p = (char*)d_ws + off;
    off += bytes;
    return p;
  };
  float*  dinv     = (float*)alloc((size_t)N_NODES * 4);
  __bf16* Xbf      = (__bf16*)alloc((size_t)N_NODES * F_IN * 2);
  __bf16* W0t      = (__bf16*)alloc((size_t)F_IN * HID * 2);
  __bf16* W1t      = (__bf16*)alloc((size_t)HID * EMB * 2);
  float*  Hbuf     = (float*)alloc((size_t)N_NODES * HID * 4);
  float*  AGG      = (float*)alloc((size_t)N_NODES * HID * 4);
  float*  paths    = (float*)alloc((size_t)BATCH * SEQ * EMB * 4);
  __bf16* paths_bf = (__bf16*)alloc((size_t)BATCH * SEQ * EMB * 2);
  float*  bn1_mu   = (float*)alloc(SEQ * 4);
  float*  bn1_rs   = (float*)alloc(SEQ * 4);
  __bf16* wih_bf   = (__bf16*)alloc((size_t)GATES * EMB * 2);
  __bf16* whh_bf   = (__bf16*)alloc((size_t)GATES * LSTM_H * 2);
  float*  hbuf     = (float*)alloc((size_t)BATCH * LSTM_H * 4);
  float*  cbuf     = (float*)alloc((size_t)BATCH * LSTM_H * 4);
  __bf16* hbfA     = (__bf16*)alloc((size_t)BATCH * LSTM_H * 2);
  __bf16* hbfB     = (__bf16*)alloc((size_t)BATCH * LSTM_H * 2);
  __bf16* hn_bf    = (__bf16*)alloc((size_t)BATCH * LSTM_H * 2);
  float*  bn2_mu   = (float*)alloc(LSTM_H * 4);
  float*  bn2_rs   = (float*)alloc(LSTM_H * 4);
  __bf16* predw_bf = (__bf16*)alloc((size_t)N_NODES * LSTM_H * 2);
  float*  logZ     = (float*)alloc(BATCH * 4);

  // ---- degree / normalization ----
  hipMemsetAsync(dinv, 0, (size_t)N_NODES * 4, stream);
  k_deg<<<(N_EDGES + 255) / 256, 256, 0, stream>>>(e_dst, dinv);
  k_dinv<<<(N_NODES + 255) / 256, 256, 0, stream>>>(dinv);

  // ---- bf16 staging of activations / weights ----
  k_f32_to_bf16<<<(N_NODES * F_IN) / 256, 256, 0, stream>>>(graph_x, Xbf,
                                                            (size_t)N_NODES * F_IN);
  k_transpose_bf16<<<(F_IN * HID + 255) / 256, 256, 0, stream>>>(w0, W0t, F_IN, HID);
  k_transpose_bf16<<<(HID * EMB + 255) / 256, 256, 0, stream>>>(w1, W1t, HID, EMB);
  k_f32_to_bf16<<<(GATES * EMB) / 256, 256, 0, stream>>>(w_ih, wih_bf,
                                                         (size_t)GATES * EMB);
  k_f32_to_bf16<<<(GATES * LSTM_H) / 256, 256, 0, stream>>>(w_hh, whh_bf,
                                                            (size_t)GATES * LSTM_H);
  k_f32_to_bf16<<<(int)(((size_t)N_NODES * LSTM_H) / 256), 256, 0, stream>>>(
      pred_w, predw_bf, (size_t)N_NODES * LSTM_H);

  // ---- GCN layer 0 ----
  k_gemm_bt<4, 0><<<dim3(N_NODES / 16, HID / 64), 32, 0, stream>>>(
      Xbf, F_IN, W0t, F_IN, nullptr, Hbuf, HID, F_IN);
  k_scale_rows<<<(N_NODES * HID) / 256, 256, 0, stream>>>(Hbuf, dinv);
  hipMemsetAsync(AGG, 0, (size_t)N_NODES * HID * 4, stream);
  k_scatter<<<(int)(((long long)N_EDGES * 32) / 256), 256, 0, stream>>>(e_src, e_dst,
                                                                        Hbuf, AGG);
  k_gcn_fin_relu_bf<<<(N_NODES * HID) / 256, 256, 0, stream>>>(AGG, Hbuf, dinv, b0, Xbf);

  // ---- GCN layer 1 -> node embeddings (f32, in Hbuf) ----
  k_gemm_bt<4, 0><<<dim3(N_NODES / 16, EMB / 64), 32, 0, stream>>>(
      Xbf, HID, W1t, HID, nullptr, Hbuf, EMB, HID);
  k_scale_rows<<<(N_NODES * EMB) / 256, 256, 0, stream>>>(Hbuf, dinv);
  hipMemsetAsync(AGG, 0, (size_t)N_NODES * EMB * 4, stream);
  k_scatter<<<(int)(((long long)N_EDGES * 32) / 256), 256, 0, stream>>>(e_src, e_dst,
                                                                        Hbuf, AGG);
  k_gcn_fin_f32<<<(N_NODES * EMB) / 256, 256, 0, stream>>>(AGG, Hbuf, dinv, b1);

  // ---- path gather + BN1 -> bf16 LSTM input ----
  k_gather<<<(BATCH * SEQ * EMB) / 256, 256, 0, stream>>>(indices, Hbuf, paths);
  k_bn1_stats<<<SEQ, 256, 0, stream>>>(paths, bn1_mu, bn1_rs);
  k_bn1_norm<<<(BATCH * SEQ * EMB) / 256, 256, 0, stream>>>(paths, bn1_mu, bn1_rs,
                                                            bn1_g, bn1_b, paths_bf);

  // ---- LSTM: 20 sequential fused WMMA steps, ping-pong bf16 h ----
  hipMemsetAsync(hbfA, 0, (size_t)BATCH * LSTM_H * 2, stream);
  hipMemsetAsync(cbuf, 0, (size_t)BATCH * LSTM_H * 4, stream);
  for (int t = 0; t < SEQ; ++t) {
    __bf16* hin  = (t & 1) ? hbfB : hbfA;
    __bf16* hout = (t & 1) ? hbfA : hbfB;
    k_lstm_step<<<dim3(BATCH / 16, LSTM_H / 16), 32, 0, stream>>>(
        t, paths_bf, wih_bf, whh_bf, b_ih, b_hh, hin, cbuf, hbuf, hout);
  }

  // ---- BN2 -> bf16 head input ----
  k_bn2_stats<<<LSTM_H, 256, 0, stream>>>(hbuf, bn2_mu, bn2_rs);
  k_bn2_norm<<<(BATCH * LSTM_H) / 256, 256, 0, stream>>>(hbuf, bn2_mu, bn2_rs, bn2_g,
                                                         bn2_b, hn_bf);

  // ---- prediction head: async-LDS staged WMMA GEMM ----
  k_gemm_head<<<dim3(BATCH / 64, N_NODES / (16 * HEAD_NT)), 128, 0, stream>>>(
      hn_bf, predw_bf, pred_b, out);

  // ---- log_softmax over 50000 classes per row ----
  k_row_logZ<<<BATCH, 256, 0, stream>>>(out, logZ);
  k_sub_logZ<<<(int)(((size_t)BATCH * N_NODES) / 256), 256, 0, stream>>>(out, logZ);
}
